// IWConLoss_13443247637324
// MI455X (gfx1250) — compile-verified, matrix operand order checked by reading, not run on hardware
//
#include <hip/hip_runtime.h>

// ---------------------------------------------------------------------------
// InfoNCE loss, fused flash-style on gfx1250 (MI455X).
//   scores = Q @ Neg^T / tau   (never materialized)
//   loss   = mean_n( logsumexp_m scores[n,:] - (q_n . p_n)/tau )
//
// Pass 1 splits the f32 inputs once into hi/lo bf16 (3xBF16 ~= FP32 trick).
// Main pass: each WG covers 64 rows (4 waves x 16-row tiles) and one quarter
// of the columns; the 16-col B tile (hi+lo bf16, 8KB) is staged in LDS with
// double-buffered global_load_async_to_lds_b128 (ASYNCcnt) and shared by all
// 4 waves -> 4x less global->L2 traffic. 3 independent WMMA accumulator
// chains (hi*hi, hi*lo, lo*hi) keep the XDL pipe busy. Online softmax runs
// in base-2 (v_exp_f32 / v_log_f32). Per-(chunk,row) (m,s) partials are
// merged by a second kernel; reductions are fixed-order (deterministic).
// ---------------------------------------------------------------------------

typedef __attribute__((ext_vector_type(16))) __bf16 v16bf;
typedef __attribute__((ext_vector_type(8)))  __bf16 v8bf;
typedef __attribute__((ext_vector_type(4)))  __bf16 v4bf;
typedef __attribute__((ext_vector_type(8)))  float  v8f;

#define TAU_INV    5.0f
#define LOG2E_TAU  7.2134752044448170f   // log2(e) / tau
#define LN2F       0.69314718055994531f
#define NQ         8192
#define NNEG       16384
#define DIM        128
#define WAVES_PB   4
#define BLOCK_T    (WAVES_PB * 32)
#define ROWS_WG    (16 * WAVES_PB)       // 64 rows per workgroup
#define ROW_BLKS   (NQ / ROWS_WG)        // 128
#define COL_TILES  (NNEG / 16)           // 1024
#define CSPLIT     4                     // column chunks
#define CT_PER     (COL_TILES / CSPLIT)  // 256 tiles per chunk
#define NBLK_MAIN  (ROW_BLKS * CSPLIT)   // 512 workgroups

#define EXP2F(x) __builtin_amdgcn_exp2f(x)   // v_exp_f32 (base-2)
#define LOG2F(x) __builtin_amdgcn_logf(x)    // v_log_f32 (base-2)

// ---- pass 1: split f32 -> (hi, lo) bf16, elementwise ----------------------
__global__ __launch_bounds__(256)
void split_prep(const float* __restrict__ src, __bf16* __restrict__ hi,
                __bf16* __restrict__ lo, int n4)
{
    int i = blockIdx.x * blockDim.x + threadIdx.x;
    if (i >= n4) return;
    float4 v = ((const float4*)src)[i];
    float x[4] = { v.x, v.y, v.z, v.w };
    v4bf h, l;
#pragma unroll
    for (int e = 0; e < 4; ++e) {
        __bf16 hh = (__bf16)x[e];
        h[e] = hh;
        l[e] = (__bf16)(x[e] - (float)hh);
    }
    ((v4bf*)hi)[i] = h;
    ((v4bf*)lo)[i] = l;
}

// Build one wave's 16-bf16 WMMA operand for K-chunk kc, lane-half h from a
// 128-bf16 row: two contiguous 8-elem (16B) groups at K = k0 and K = k0+16.
__device__ __forceinline__ v16bf load_chunk(const __bf16* __restrict__ row,
                                            int kc, int h)
{
    const v8bf* r8 = (const v8bf*)row;
    int g = kc * 4 + h;                 // (kc*32 + 8h) / 8
    v8bf a = r8[g];
    v8bf b = r8[g + 2];
    return __builtin_shufflevector(a, b,
        0, 1, 2, 3, 4, 5, 6, 7, 8, 9, 10, 11, 12, 13, 14, 15);
}

// Async-copy one 16x128 bf16 tile pair (hi 4KB + lo 4KB) into LDS.
// 128 threads x 16B x 2 offsets per part; tracked by ASYNCcnt.
__device__ __forceinline__ void async_tile(const __bf16* __restrict__ nh,
                                           const __bf16* __restrict__ nl,
                                           int t, __bf16* lds_base, int tid)
{
    unsigned lh = (unsigned)(uintptr_t)lds_base + (unsigned)tid * 16u;
    unsigned ll = lh + 4096u;
    unsigned long long gh = (unsigned long long)(uintptr_t)nh
                          + (unsigned long long)t * 4096ull + (unsigned long long)tid * 16ull;
    unsigned long long gl = (unsigned long long)(uintptr_t)nl
                          + (unsigned long long)t * 4096ull + (unsigned long long)tid * 16ull;
    asm volatile(
        "global_load_async_to_lds_b128 %0, %2, off\n\t"
        "global_load_async_to_lds_b128 %0, %2, off offset:2048\n\t"
        "global_load_async_to_lds_b128 %1, %3, off\n\t"
        "global_load_async_to_lds_b128 %1, %3, off offset:2048"
        :
        : "v"(lh), "v"(ll), "v"(gh), "v"(gl)
        : "memory");
}

__device__ __forceinline__ void wait_async0()
{
    asm volatile("s_wait_asynccnt 0x0" ::: "memory");
}

// ---- pass 2: fused GEMM + online logsumexp --------------------------------
__global__ __launch_bounds__(BLOCK_T)
void iwcon_main(const __bf16* __restrict__ qh, const __bf16* __restrict__ ql,
                const __bf16* __restrict__ nh, const __bf16* __restrict__ nl,
                float* __restrict__ mpart, float* __restrict__ spart)
{
    __shared__ __bf16 sbuf[2][2][16][DIM];   // [buf][hi/lo][col][k] = 16KB

    const int tid  = threadIdx.x;
    const int lane = tid & 31;
    const int wave = tid >> 5;
    const int m16  = lane & 15;      // A: row-in-tile / B: col-in-tile
    const int h    = lane >> 4;      // lane half selects K sub-pattern

    const int rowBlk = blockIdx.x & (ROW_BLKS - 1);
    const int chunk  = blockIdx.x / ROW_BLKS;
    const int r0     = rowBlk * ROWS_WG + wave * 16;   // this wave's 16 rows
    const int t0     = chunk * CT_PER;

    // A tile (16 queries x 128) hi/lo bf16, resident in VGPRs
    v16bf a_hi[4], a_lo[4];
    {
        const __bf16* qhr = qh + (size_t)(r0 + m16) * DIM;
        const __bf16* qlr = ql + (size_t)(r0 + m16) * DIM;
#pragma unroll
        for (int kc = 0; kc < 4; ++kc) {
            a_hi[kc] = load_chunk(qhr, kc, h);
            a_lo[kc] = load_chunk(qlr, kc, h);
        }
    }

    // online softmax state (base-2): 8 rows per lane (row = j + 8*h)
    float mrun[8], srun[8];
#pragma unroll
    for (int j = 0; j < 8; ++j) { mrun[j] = -__builtin_inff(); srun[j] = 0.0f; }

    // prologue: stage first tile into buffer 0
    async_tile(nh, nl, t0, &sbuf[0][0][0][0], tid);
    wait_async0();
    __syncthreads();

    for (int i = 0; i < CT_PER; ++i) {
        const int p = i & 1;
        if (i + 1 < CT_PER)
            async_tile(nh, nl, t0 + i + 1, &sbuf[p ^ 1][0][0][0], tid);

        const __bf16* bh = &sbuf[p][0][m16][0];
        const __bf16* bl = &sbuf[p][1][m16][0];

        v8f acc0 = {}, acc1 = {}, acc2 = {};     // 3 independent XDL chains
#pragma unroll
        for (int kc = 0; kc < 4; ++kc) {
            v16bf b_hi = load_chunk(bh, kc, h);
            v16bf b_lo = load_chunk(bl, kc, h);
            acc0 = __builtin_amdgcn_wmma_f32_16x16x32_bf16(
                       false, a_hi[kc], false, b_hi, (short)0, acc0, false, false);
            acc1 = __builtin_amdgcn_wmma_f32_16x16x32_bf16(
                       false, a_hi[kc], false, b_lo, (short)0, acc1, false, false);
            acc2 = __builtin_amdgcn_wmma_f32_16x16x32_bf16(
                       false, a_lo[kc], false, b_hi, (short)0, acc2, false, false);
        }
#pragma unroll
        for (int j = 0; j < 8; ++j) {
            float sc = (acc0[j] + acc1[j] + acc2[j]) * LOG2E_TAU;
            float mo = mrun[j];
            float mn = fmaxf(mo, sc);
            srun[j]  = srun[j] * EXP2F(mo - mn) + EXP2F(sc - mn);
            mrun[j]  = mn;
        }

        wait_async0();       // next buffer resident (this wave's share)
        __syncthreads();     // all waves done loading next + reading current
    }

    // combine across the 16 lanes of each half (columns)
#pragma unroll
    for (int off = 1; off < 16; off <<= 1) {
#pragma unroll
        for (int j = 0; j < 8; ++j) {
            float mo = __shfl_xor(mrun[j], off, 32);
            float so = __shfl_xor(srun[j], off, 32);
            float mn = fmaxf(mrun[j], mo);
            srun[j]  = srun[j] * EXP2F(mrun[j] - mn) + so * EXP2F(mo - mn);
            mrun[j]  = mn;
        }
    }

    // each wave owns distinct rows: write per-(chunk,row) partials directly
    if (m16 == 0) {
#pragma unroll
        for (int j = 0; j < 8; ++j) {
            int row = r0 + j + 8 * h;
            mpart[(size_t)chunk * NQ + row] = mrun[j];
            spart[(size_t)chunk * NQ + row] = srun[j];
        }
    }
}

// ---- pass 3: merge column-chunk partials, add positive logit, reduce ------
__global__ __launch_bounds__(256)
void iwcon_merge(const float* __restrict__ q, const float* __restrict__ pos,
                 const float* __restrict__ mpart, const float* __restrict__ spart,
                 float* __restrict__ blocksum)
{
    int r = blockIdx.x * 256 + threadIdx.x;
    float mm = mpart[r], ss = spart[r];
#pragma unroll
    for (int c = 1; c < CSPLIT; ++c) {
        float mo = mpart[(size_t)c * NQ + r], so = spart[(size_t)c * NQ + r];
        float mn = fmaxf(mm, mo);
        ss = ss * EXP2F(mm - mn) + so * EXP2F(mo - mn);
        mm = mn;
    }
    float lse = (mm + LOG2F(ss)) * LN2F;       // back to natural log

    const float4* q4 = (const float4*)(q   + (size_t)r * DIM);
    const float4* p4 = (const float4*)(pos + (size_t)r * DIM);
    float dot = 0.0f;
#pragma unroll
    for (int i = 0; i < DIM / 4; ++i) {
        float4 a = q4[i], b = p4[i];
        dot += a.x * b.x + a.y * b.y + a.z * b.z + a.w * b.w;
    }
    float val = lse - dot * TAU_INV;

    __shared__ float sm[256];
    sm[threadIdx.x] = val;
    __syncthreads();
    for (int s = 128; s > 0; s >>= 1) {
        if (threadIdx.x < s) sm[threadIdx.x] += sm[threadIdx.x + s];
        __syncthreads();
    }
    if (threadIdx.x == 0) blocksum[blockIdx.x] = sm[0];
}

__global__ __launch_bounds__(32)
void iwcon_finish(const float* __restrict__ blocksum, float* __restrict__ out)
{
    float v = blocksum[threadIdx.x];           // 32 partials, one wave
#pragma unroll
    for (int off = 16; off > 0; off >>= 1) v += __shfl_xor(v, off, 32);
    if (threadIdx.x == 0) out[0] = v * (1.0f / NQ);
}

extern "C" void kernel_launch(void* const* d_in, const int* in_sizes, int n_in,
                              void* d_out, int out_size, void* d_ws, size_t ws_size,
                              hipStream_t stream)
{
    const float* q   = (const float*)d_in[0];   // [8192, 128]
    const float* pos = (const float*)d_in[1];   // [8192, 128]
    const float* neg = (const float*)d_in[2];   // [16384, 128]
    float* out       = (float*)d_out;           // scalar

    // workspace layout (~12.3 MB)
    char* ws = (char*)d_ws;
    const size_t QB = (size_t)NQ * DIM * sizeof(__bf16);     // 2 MB
    const size_t NB = (size_t)NNEG * DIM * sizeof(__bf16);   // 4 MB
    const size_t PB = (size_t)CSPLIT * NQ * sizeof(float);   // 128 KB
    __bf16* qh = (__bf16*)(ws);
    __bf16* ql = (__bf16*)(ws + QB);
    __bf16* nh = (__bf16*)(ws + 2 * QB);
    __bf16* nl = (__bf16*)(ws + 2 * QB + NB);
    float* mpart    = (float*)(ws + 2 * QB + 2 * NB);
    float* spart    = (float*)(ws + 2 * QB + 2 * NB + PB);
    float* blocksum = (float*)(ws + 2 * QB + 2 * NB + 2 * PB);

    const int q4 = NQ * DIM / 4;     // 262144 float4s
    const int n4 = NNEG * DIM / 4;   // 524288 float4s
    split_prep<<<q4 / 256, 256, 0, stream>>>(q,   qh, ql, q4);
    split_prep<<<n4 / 256, 256, 0, stream>>>(neg, nh, nl, n4);

    iwcon_main<<<NBLK_MAIN, BLOCK_T, 0, stream>>>(qh, ql, nh, nl, mpart, spart);
    iwcon_merge<<<NQ / 256, 256, 0, stream>>>(q, pos, mpart, spart, blocksum);
    iwcon_finish<<<1, 32, 0, stream>>>(blocksum, out);
}
